// model_4080218931288
// MI455X (gfx1250) — matrix operand order repaired and checked by reading, not compiled, not run on hardware
//
#include <hip/hip_runtime.h>
#include <hip/hip_bf16.h>

typedef __attribute__((ext_vector_type(16))) __bf16 v16bf;
typedef __attribute__((ext_vector_type(8)))  __bf16 v8bf;
typedef __attribute__((ext_vector_type(8)))  float  v8f;

#define Bn    1024
#define TXn   32
#define TVn   64
#define Fn    128
#define Hn    1024
#define PREDn 64
#define KTOT  1152          // F + H
#define KC_G  36            // KTOT/32
#define KC_R  32            // H/32
#define AST   1160          // padded LDS row stride (bf16 elems): 2320B -> 4-bank skew, conflict-free b128

// workspace layout (bytes)
#define WC_BYTES   (4096u * 1152u * 2u)          // 9,437,184
#define WR_BYTES   (128u * 1024u * 2u)           //   262,144
#define BIAS_OFF   (WC_BYTES + WR_BYTES)

__device__ __forceinline__ float sigm_f(float x)  { return 1.0f / (1.0f + __expf(-x)); }
__device__ __forceinline__ float tanh_f(float x)  { return 2.0f / (1.0f + __expf(-2.0f * x)) - 1.0f; }

__device__ __forceinline__ v16bf load_afrag(const __bf16* p) {
    // A 16x32 bf16 fragment, per ISA: lane row = lane%16,
    // K = {base..base+7} U {base+16..base+23}, base = (lane/16)*8
    v8bf lo = *(const v8bf*)(p);
    v8bf hi = *(const v8bf*)(p + 16);
    v16bf a;
#pragma unroll
    for (int i = 0; i < 8; ++i) { a[i] = lo[i]; a[i + 8] = hi[i]; }
    return a;
}

// ---- prep: pack [W_ih | W_hh] (4096 x 1152) into bf16 B-fragments ------------
// frag(nt,kc): 512 bf16, lane l gets 16 contiguous: col j = nt*16 + l%16,
// K = kc*32 + (l/16)*8 + (p&7) + (p>>3)*16   (p = 0..15)
__global__ void pack_wc_kernel(const float* __restrict__ Wih,
                               const float* __restrict__ Whh,
                               __bf16* __restrict__ Wc) {
    unsigned i = blockIdx.x * 256u + threadIdx.x;      // < 4096*1152
    int p     = i & 15;
    int lanep = (i >> 4) & 31;
    unsigned frag = i >> 9;
    int kc = frag % KC_G;
    int nt = frag / KC_G;                               // 0..255 (4096/16 tiles)
    int j  = nt * 16 + (lanep & 15);
    int k  = kc * 32 + (lanep >> 4) * 8 + (p & 7) + (p >> 3) * 16;
    float val = (k < Fn) ? Wih[(size_t)j * Fn + k]
                         : Whh[(size_t)j * Hn + (k - Fn)];
    Wc[i] = (__bf16)val;
}

// ---- prep: pack W_reg (128 x 1024) into bf16 B-fragments ---------------------
__global__ void pack_wr_kernel(const float* __restrict__ Wreg,
                               __bf16* __restrict__ Wr) {
    unsigned i = blockIdx.x * 256u + threadIdx.x;      // < 128*1024
    int p     = i & 15;
    int lanep = (i >> 4) & 31;
    unsigned frag = i >> 9;
    int kc = frag % KC_R;
    int nt = frag / KC_R;                               // 0..7
    int j  = nt * 16 + (lanep & 15);
    int k  = kc * 32 + (lanep >> 4) * 8 + (p & 7) + (p >> 3) * 16;
    Wr[i] = (__bf16)Wreg[(size_t)j * Hn + k];
}

__global__ void bias_kernel(const float* __restrict__ bih,
                            const float* __restrict__ bhh,
                            float* __restrict__ bias) {
    int i = blockIdx.x * 256 + threadIdx.x;             // < 4096
    bias[i] = bih[i] + bhh[i];
}

// ---- y_out = transpose(y, (1,0,2)) ------------------------------------------
__global__ void transpose_y_kernel(const float* __restrict__ y,
                                   float* __restrict__ yo) {
    size_t i = (size_t)blockIdx.x * 256u + threadIdx.x; // < PRED*B*F
    int f = (int)(i % Fn);
    size_t r = i / Fn;
    int b = (int)(r % Bn);
    int pidx = (int)(r / Bn);
    yo[i] = y[((size_t)b * PREDn + pidx) * Fn + f];
}

// ---- main recurrent kernel: one block = 16 batch rows, 8 waves --------------
__global__ __launch_bounds__(256, 1) void lstm_forecast_kernel(
    const float*  __restrict__ x,
    const float*  __restrict__ v,
    const float*  __restrict__ breg,
    const __bf16* __restrict__ Wc,
    const __bf16* __restrict__ Wr,
    const float*  __restrict__ bias,
    float*        __restrict__ out0)
{
    __shared__ __bf16 Abuf[2][16 * AST];   // double-buffered [v_t | h] panel, 16 x 1152 bf16

    const int tid  = threadIdx.x;
    const int wave = tid >> 5;
    const int lane = tid & 31;
    const int n    = lane & 15;            // C-fragment column, also A row
    const int half = lane >> 4;            // 0: M/K low half, 1: high half
    const int koff = half * 8;
    const int b0   = blockIdx.x * 16;

    // zero the h region of buffer 0 (h0 = 0)
    for (int i = tid; i < 16 * Hn; i += 256) {
        int rr = i >> 10, cc = i & (Hn - 1);
        Abuf[0][rr * AST + Fn + cc] = (__bf16)0.0f;
    }

    v8f cst[8];                            // cell state: 8 n-tiles per wave, fp32
#pragma unroll
    for (int jt = 0; jt < 8; ++jt) cst[jt] = {};
    v8f xs = {};                           // running forecast x (this wave's 16 cols)

    const size_t GATE_STRIDE = (size_t)64 * KC_G * 512; // frag offset between gates

    for (int t = 0; t < 127; ++t) {
        const int cur = t & 1, nxt = cur ^ 1;

        // stage encoder input v[:, t, :] into Abuf[cur][:, 0:F] as bf16
        if (t < TVn) {
#pragma unroll
            for (int i = 0; i < 8; ++i) {
                int e   = tid * 8 + i;       // 0..2047
                int rr  = e >> 7;
                int cc  = e & (Fn - 1);
                Abuf[cur][rr * AST + cc] =
                    (__bf16)v[((size_t)(b0 + rr) * TVn + t) * Fn + cc];
            }
        }
        __syncthreads();   // input + previous step's writes visible

        // ---- gates = [v_t | h] @ Wc.T, this wave owns H-cols [wave*128, +128) ----
        const __bf16* arow = &Abuf[cur][n * AST + koff];
#pragma unroll
        for (int jt = 0; jt < 8; ++jt) {
            const int j0 = wave * 128 + jt * 16;           // column within a gate
            const __bf16* wbase = Wc + ((size_t)(j0 >> 4) * KC_G) * 512 + (size_t)lane * 16;

            v8f ai = {}, af = {}, ag = {}, ao = {};
            for (int kc = 0; kc < KC_G; ++kc) {
                v16bf a = load_afrag(arow + kc * 32);
                const __bf16* wp = wbase + (size_t)kc * 512;
                if (kc + 1 < KC_G)
                    __builtin_prefetch((const void*)(wp + 512), 0, 0);
                v16bf bI = *(const v16bf*)(wp);
                v16bf bF = *(const v16bf*)(wp + GATE_STRIDE);
                v16bf bG = *(const v16bf*)(wp + 2 * GATE_STRIDE);
                v16bf bO = *(const v16bf*)(wp + 3 * GATE_STRIDE);
                ai = __builtin_amdgcn_wmma_f32_16x16x32_bf16(false, a, false, bI, (short)0, ai, false, false);
                af = __builtin_amdgcn_wmma_f32_16x16x32_bf16(false, a, false, bF, (short)0, af, false, false);
                ag = __builtin_amdgcn_wmma_f32_16x16x32_bf16(false, a, false, bG, (short)0, ag, false, false);
                ao = __builtin_amdgcn_wmma_f32_16x16x32_bf16(false, a, false, bO, (short)0, ao, false, false);
            }

            const float bI_ = bias[0 * Hn + j0 + n];
            const float bF_ = bias[1 * Hn + j0 + n];
            const float bG_ = bias[2 * Hn + j0 + n];
            const float bO_ = bias[3 * Hn + j0 + n];
            v8f cv = cst[jt];
#pragma unroll
            for (int r = 0; r < 8; ++r) {
                float I = sigm_f(ai[r] + bI_);
                float Fg = sigm_f(af[r] + bF_);
                float G = tanh_f(ag[r] + bG_);
                float O = sigm_f(ao[r] + bO_);
                float cc = Fg * cv[r] + I * G;
                cv[r] = cc;
                float hh = O * tanh_f(cc);
                int rrow = r + half * 8;                    // C layout: M = r + 8*(lane/16)
                Abuf[nxt][rrow * AST + Fn + j0 + n] = (__bf16)hh;
            }
            cst[jt] = cv;
        }
        __syncthreads();   // all of h_t written to Abuf[nxt]

        // ---- regression head + forecast accumulation (steps 63..126) ----
        if (t >= 63) {
            const int col = wave * 16 + n;                  // wave owns 16 of F cols
            const __bf16* hrow = &Abuf[nxt][n * AST + Fn + koff];
            const __bf16* wrb  = Wr + (size_t)wave * KC_R * 512 + (size_t)lane * 16;
            v8f acc = {};
            for (int kc = 0; kc < KC_R; ++kc) {
                v16bf a = load_afrag(hrow + kc * 32);
                v16bf b = *(const v16bf*)(wrb + (size_t)kc * 512);
                acc = __builtin_amdgcn_wmma_f32_16x16x32_bf16(false, a, false, b, (short)0, acc, false, false);
            }
            const float bv = breg[col];
#pragma unroll
            for (int r = 0; r < 8; ++r) {
                int rrow = r + half * 8;
                float vn = acc[r] + bv;
                float xn;
                if (t == 63)
                    xn = x[((size_t)(b0 + rrow) * TXn + (TXn - 1)) * Fn + col] + vn;
                else
                    xn = xs[r] + vn;
                xs[r] = xn;
                out0[((size_t)(t - 63) * Bn + b0 + rrow) * Fn + col] = xn;
                if (t < 126)
                    Abuf[nxt][rrow * AST + col] = (__bf16)vn;   // decoder input for t+1
            }
        }
    }
}

extern "C" void kernel_launch(void* const* d_in, const int* in_sizes, int n_in,
                              void* d_out, int out_size, void* d_ws, size_t ws_size,
                              hipStream_t stream) {
    const float* x     = (const float*)d_in[0];
    const float* y     = (const float*)d_in[1];
    const float* v     = (const float*)d_in[2];
    const float* W_ih  = (const float*)d_in[3];
    const float* W_hh  = (const float*)d_in[4];
    const float* b_ih  = (const float*)d_in[5];
    const float* b_hh  = (const float*)d_in[6];
    const float* W_reg = (const float*)d_in[7];
    const float* b_reg = (const float*)d_in[8];
    float* out = (float*)d_out;

    char* ws = (char*)d_ws;
    __bf16* Wc   = (__bf16*)ws;                       // 4096x1152 bf16 fragments
    __bf16* Wr   = (__bf16*)(ws + WC_BYTES);          // 128x1024 bf16 fragments
    float*  bias = (float*)(ws + BIAS_OFF);           // 4096 fp32

    // prep
    pack_wc_kernel<<<(4096u * 1152u) / 256u, 256, 0, stream>>>(W_ih, W_hh, Wc);
    pack_wr_kernel<<<(128u * 1024u) / 256u, 256, 0, stream>>>(W_reg, Wr);
    bias_kernel<<<4096 / 256, 256, 0, stream>>>(b_ih, b_hh, bias);

    // y_out = transpose(y)
    const size_t OUTE = (size_t)PREDn * Bn * Fn;      // 8,388,608
    transpose_y_kernel<<<(unsigned)(OUTE / 256), 256, 0, stream>>>(y, out + OUTE);

    // recurrent forecast
    lstm_forecast_kernel<<<Bn / 16, 256, 0, stream>>>(x, v, b_reg, Wc, Wr, bias, out);
}
